// RQVAE_17257178595882
// MI455X (gfx1250) — matrix operand order repaired and checked by reading, not compiled, not run on hardware
//
#include <hip/hip_runtime.h>
#include <stdint.h>

// ---------------------------------------------------------------------------
// Types for CDNA5 WMMA (wave32, v_wmma_f32_16x16x32_bf16)
// ---------------------------------------------------------------------------
typedef __attribute__((ext_vector_type(16))) __bf16 v16bf;
typedef __attribute__((ext_vector_type(8)))  __bf16 v8bf;
typedef __attribute__((ext_vector_type(8)))  float  v8f;

#define NROWS   131072
#define INDIM   768
#define EDIM    64
#define NLEVELS 4
#define NCODES  256
#define MU      0.25f

// f32 -> bf16 RNE (integer path)
static __device__ __forceinline__ __bf16 f2bf(float f) {
    union { float f; uint32_t u; } v; v.f = f;
    uint32_t r = (v.u + 0x7FFFu + ((v.u >> 16) & 1u)) >> 16;
    union { unsigned short s; __bf16 b; } o; o.s = (unsigned short)r;
    return o.b;
}
static __device__ __forceinline__ uint32_t f2bf_pack2(float a, float b) {
    union { float f; uint32_t u; } x, y; x.f = a; y.f = b;
    uint32_t ra = (x.u + 0x7FFFu + ((x.u >> 16) & 1u)) >> 16;
    uint32_t rb = (y.u + 0x7FFFu + ((y.u >> 16) & 1u)) >> 16;
    return (rb << 16) | (ra & 0xFFFFu);
}

// ---------------------------------------------------------------------------
// Prep kernels
// ---------------------------------------------------------------------------
// f32 (K x N) row-major -> bf16 (N x K) (transposed, GEMM-B layout)
__global__ void k_transpose_bf16(const float* __restrict__ in,
                                 __bf16* __restrict__ out, int K, int N) {
    int total = K * N;
    for (int t = blockIdx.x * blockDim.x + threadIdx.x; t < total;
         t += gridDim.x * blockDim.x) {
        int n = t / K, k = t - n * K;
        out[t] = f2bf(in[(size_t)k * N + n]);
    }
}

__global__ void k_f32_to_bf16(const float* __restrict__ in,
                              __bf16* __restrict__ out, int total) {
    for (int t = blockIdx.x * blockDim.x + threadIdx.x; t < total;
         t += gridDim.x * blockDim.x)
        out[t] = f2bf(in[t]);
}

__global__ void k_code_norms(const float* __restrict__ cb,
                             float* __restrict__ norms) {
    int c = blockIdx.x * blockDim.x + threadIdx.x;
    if (c < NLEVELS * NCODES) {
        float s = 0.f;
        #pragma unroll
        for (int d = 0; d < EDIM; ++d) { float v = cb[(size_t)c * EDIM + d]; s += v * v; }
        norms[c] = s;
    }
}

__global__ void k_zero_f32(float* __restrict__ p, int total) {
    for (int t = blockIdx.x * blockDim.x + threadIdx.x; t < total;
         t += gridDim.x * blockDim.x)
        p[t] = 0.f;
}

// ---------------------------------------------------------------------------
// bf16-WMMA GEMM:  C[M,N] = act(alpha * (A[M,K] @ Bt[N,K]^T) + bias[N])
//   BN    : block tile N (64 or 128); block tile M = 128, BK = 32
//   ABF   : A is bf16 (uint4 copy) vs f32 (convert while staging)
//   CMODE : 0 = f32 out, 1 = bf16 out, 2 = both (f32 + bf16 copies)
// 256 threads = 8 waves; wave tile 32 x (BN/2); register-double-buffered
// staging so next k-tile global loads overlap the WMMAs.
// All dims here are multiples of the tile dims -> no edge handling.
// ---------------------------------------------------------------------------
template <int BN, bool ABF, int CMODE>
__global__ __launch_bounds__(256)
void k_gemm(const void* __restrict__ Av, const __bf16* __restrict__ Bt,
            const float* __restrict__ bias, float* __restrict__ Cf,
            __bf16* __restrict__ Cb, int M, int N, int K, float alpha, int relu) {
    constexpr int JT = BN / 32;   // B fragments per wave
    constexpr int BL = BN / 64;   // uint4 B loads per thread per tile
    __shared__ __align__(16) __bf16 sA[128 * 48];
    __shared__ __align__(16) __bf16 sB[BN * 48];

    const int t = threadIdx.x, lane = t & 31, wave = t >> 5;
    const int wr = wave >> 1, wc = wave & 1;
    const int mBase = blockIdx.y * 128;
    const int nBase = blockIdx.x * BN;

    const float*  Af = (const float*)Av;
    const __bf16* Ab = (const __bf16*)Av;

    uint4 ra[2]; float4 rf[4]; uint4 rb[BL];

    auto loadTiles = [&](int kb) {
        if constexpr (ABF) {
            #pragma unroll
            for (int i = 0; i < 2; ++i) {      // 128x32 bf16 = 512 uint4
                int u = i * 256 + t;
                int row = u >> 2, k8 = (u & 3) * 8;
                ra[i] = *reinterpret_cast<const uint4*>(
                    &Ab[(size_t)(mBase + row) * K + kb + k8]);
            }
        } else {
            #pragma unroll
            for (int i = 0; i < 4; ++i) {      // 128x32 f32 = 1024 float4
                int f = i * 256 + t;
                int row = f >> 3, kv = (f & 7) * 4;
                rf[i] = *reinterpret_cast<const float4*>(
                    &Af[(size_t)(mBase + row) * K + kb + kv]);
            }
        }
        #pragma unroll
        for (int j = 0; j < BL; ++j) {         // BN x 32 bf16
            int u = j * 256 + t;
            int col = u >> 2, k8 = (u & 3) * 8;
            rb[j] = *reinterpret_cast<const uint4*>(
                &Bt[(size_t)(nBase + col) * K + kb + k8]);
        }
    };
    auto storeTiles = [&]() {
        if constexpr (ABF) {
            #pragma unroll
            for (int i = 0; i < 2; ++i) {
                int u = i * 256 + t;
                int row = u >> 2, k8 = (u & 3) * 8;
                *reinterpret_cast<uint4*>(&sA[row * 48 + k8]) = ra[i];
            }
        } else {
            #pragma unroll
            for (int i = 0; i < 4; ++i) {
                int f = i * 256 + t;
                int row = f >> 3, kv = (f & 7) * 4;
                uint2 p;
                p.x = f2bf_pack2(rf[i].x, rf[i].y);
                p.y = f2bf_pack2(rf[i].z, rf[i].w);
                *reinterpret_cast<uint2*>(&sA[row * 48 + kv]) = p;
            }
        }
        #pragma unroll
        for (int j = 0; j < BL; ++j) {
            int u = j * 256 + t;
            int col = u >> 2, k8 = (u & 3) * 8;
            *reinterpret_cast<uint4*>(&sB[col * 48 + k8]) = rb[j];
        }
    };

    v8f acc[2][JT] = {};
    loadTiles(0);
    for (int kb = 0; kb < K; kb += 32) {
        __syncthreads();          // prior compute done reading LDS
        storeTiles();
        __syncthreads();
        if (kb + 32 < K) loadTiles(kb + 32);   // in flight during WMMAs

        const int r = lane & 15, hi = lane >> 4, klo = hi * 8;
        v16bf afrag[2], bfrag[JT];
        #pragma unroll
        for (int i = 0; i < 2; ++i) {  // A: lanes0-15 K{0..7,16..23}; 16-31 +8
            const int row = wr * 32 + i * 16 + r;
            const v8bf a0 = *reinterpret_cast<const v8bf*>(&sA[row * 48 + klo]);
            const v8bf a1 = *reinterpret_cast<const v8bf*>(&sA[row * 48 + 16 + klo]);
            #pragma unroll
            for (int e = 0; e < 8; ++e) { afrag[i][e] = a0[e]; afrag[i][e + 8] = a1[e]; }
        }
        #pragma unroll
        for (int j = 0; j < JT; ++j) { // B: lane holds col lane&15, K{0..7,16..23} + hi*8
            const int col = wc * (BN / 2) + j * 16 + r;
            const int koff = hi * 8;
            const v8bf b0 = *reinterpret_cast<const v8bf*>(&sB[col * 48 + koff]);
            const v8bf b1 = *reinterpret_cast<const v8bf*>(&sB[col * 48 + koff + 16]);
            #pragma unroll
            for (int e = 0; e < 8; ++e) { bfrag[j][e] = b0[e]; bfrag[j][e + 8] = b1[e]; }
        }
        #pragma unroll
        for (int i = 0; i < 2; ++i)
            #pragma unroll
            for (int j = 0; j < JT; ++j)
                acc[i][j] = __builtin_amdgcn_wmma_f32_16x16x32_bf16(
                    false, afrag[i], false, bfrag[j], (short)0, acc[i][j],
                    false, false);
    }

    // epilogue: C/D layout -> row = v + 8*(lane>>4), col = lane&15
    #pragma unroll
    for (int i = 0; i < 2; ++i) {
        #pragma unroll
        for (int j = 0; j < JT; ++j) {
            const int col = nBase + wc * (BN / 2) + j * 16 + (lane & 15);
            const float b = bias ? bias[col] : 0.f;
            #pragma unroll
            for (int v = 0; v < 8; ++v) {
                const int row = mBase + wr * 32 + i * 16 + (lane >> 4) * 8 + v;
                float val = alpha * acc[i][j][v] + b;
                if (relu) val = fmaxf(val, 0.f);
                const size_t o = (size_t)row * N + col;
                if constexpr (CMODE == 0) { Cf[o] = val; }
                else if constexpr (CMODE == 1) { Cb[o] = f2bf(val); }
                else { Cf[o] = val; Cb[o] = f2bf(val); }
            }
        }
    }
}

// ---------------------------------------------------------------------------
// RQ per-level update. One wave per row.
// S[row,256] = |emb_j|^2 - 2*res.emb_j  (row-constant |res|^2 omitted)
// ---------------------------------------------------------------------------
__global__ __launch_bounds__(256)
void k_rq_update(const float* __restrict__ S, const float* __restrict__ cb,
                 float* __restrict__ res, __bf16* __restrict__ res_bf,
                 float* __restrict__ xq, float* __restrict__ lossAcc,
                 float* __restrict__ idxOut, int level) {
    const int lane = threadIdx.x & 31;
    const int wave = threadIdx.x >> 5;
    const int row  = blockIdx.x * 8 + wave;

    const float4* s4 = reinterpret_cast<const float4*>(&S[(size_t)row * NCODES]);
    const float4 v0 = s4[lane * 2], v1 = s4[lane * 2 + 1];
    const float vals[8] = { v0.x, v0.y, v0.z, v0.w, v1.x, v1.y, v1.z, v1.w };
    float minv = 3.4e38f; int mini = 0;
    #pragma unroll
    for (int j = 0; j < 8; ++j) {
        if (vals[j] < minv) { minv = vals[j]; mini = lane * 8 + j; }
    }
    #pragma unroll
    for (int off = 16; off >= 1; off >>= 1) {   // first-index tie-break
        const float ov = __shfl_xor(minv, off, 32);
        const int   oi = __shfl_xor(mini, off, 32);
        if (ov < minv || (ov == minv && oi < mini)) { minv = ov; mini = oi; }
    }

    const float* q = &cb[(size_t)mini * EDIM];
    float sse = 0.f;
    #pragma unroll
    for (int p = 0; p < 2; ++p) {
        const int d = lane + p * 32;
        const float qv = q[d];
        const size_t o = (size_t)row * EDIM + d;
        const float r = res[o];
        const float diff = qv - r;
        sse += diff * diff;
        xq[o] += qv;               // straight-through: x_q accumulates q
        const float nr = r - qv;   // residual for next level
        res[o] = nr;
        res_bf[o] = f2bf(nr);      // bf16 copy feeds next distance GEMM
    }
    #pragma unroll
    for (int off = 16; off >= 1; off >>= 1) sse += __shfl_xor(sse, off, 32);
    if (lane == 0) {
        atomicAdd(&lossAcc[level], sse);
        idxOut[(size_t)row * NLEVELS + level] = (float)mini;
    }
}

__global__ void k_finalize_loss(const float* __restrict__ lossAcc,
                                float* __restrict__ out) {
    if (threadIdx.x == 0 && blockIdx.x == 0) {
        float s = lossAcc[0] + lossAcc[1] + lossAcc[2] + lossAcc[3];
        out[0] = s * ((1.f + MU) / ((float)NLEVELS * (float)NROWS * (float)EDIM));
    }
}

// ---------------------------------------------------------------------------
// Host-side orchestration
// ---------------------------------------------------------------------------
extern "C" void kernel_launch(void* const* d_in, const int* in_sizes, int n_in,
                              void* d_out, int out_size, void* d_ws, size_t ws_size,
                              hipStream_t stream) {
    (void)in_sizes; (void)n_in; (void)out_size; (void)ws_size;

    const float* x   = (const float*)d_in[0];
    const float* We0 = (const float*)d_in[1];
    const float* be0 = (const float*)d_in[2];
    const float* We1 = (const float*)d_in[3];
    const float* be1 = (const float*)d_in[4];
    const float* We2 = (const float*)d_in[5];
    const float* be2 = (const float*)d_in[6];
    const float* Wd0 = (const float*)d_in[7];
    const float* bd0 = (const float*)d_in[8];
    const float* Wd1 = (const float*)d_in[9];
    const float* bd1 = (const float*)d_in[10];
    const float* Wd2 = (const float*)d_in[11];
    const float* bd2 = (const float*)d_in[12];
    const float* cb  = (const float*)d_in[13];

    float* out = (float*)d_out;
    float* out_loss = out + (size_t)NROWS * INDIM;
    float* out_idx  = out_loss + 1;

    // ---- workspace layout (offsets multiple of 256 B)
    char* ws = (char*)d_ws;
    __bf16* act1_bf = (__bf16*)(ws + 0ull);            // 131072x512 bf16
    __bf16* act2_bf = (__bf16*)(ws + 134217728ull);    // 131072x256 bf16
    float*  res     = (float*)(ws + 201326592ull);     // 131072x64 f32
    __bf16* res_bf  = (__bf16*)(ws + 234881024ull);    // 131072x64 bf16
    float*  xq      = (float*)(ws + 251658240ull);     // 131072x64 f32
    float*  lossAcc = (float*)(ws + 285212672ull);     // 4 (pad 256), adj to xq
    __bf16* xq_bf   = (__bf16*)(ws + 285212928ull);    // 131072x64 bf16
    float*  S       = (float*)(ws + 301990144ull);     // 131072x256 f32
    __bf16* wt_e0   = (__bf16*)(ws + 436207872ull);    // 512x768
    __bf16* wt_e1   = (__bf16*)(ws + 436994304ull);    // 256x512
    __bf16* wt_e2   = (__bf16*)(ws + 437256448ull);    // 64x256
    __bf16* wt_d0   = (__bf16*)(ws + 437289216ull);    // 256x64
    __bf16* wt_d1   = (__bf16*)(ws + 437321984ull);    // 512x256
    __bf16* wt_d2   = (__bf16*)(ws + 437584128ull);    // 768x512
    __bf16* cb_bf   = (__bf16*)(ws + 438370560ull);    // 4x256x64
    float*  norms   = (float*)(ws + 438501632ull);     // 4x256

    const dim3 blk(256);

    // ---- weight prep
    k_transpose_bf16<<<dim3(1536), blk, 0, stream>>>(We0, wt_e0, 768, 512);
    k_transpose_bf16<<<dim3(512),  blk, 0, stream>>>(We1, wt_e1, 512, 256);
    k_transpose_bf16<<<dim3(64),   blk, 0, stream>>>(We2, wt_e2, 256, 64);
    k_transpose_bf16<<<dim3(64),   blk, 0, stream>>>(Wd0, wt_d0, 64, 256);
    k_transpose_bf16<<<dim3(512),  blk, 0, stream>>>(Wd1, wt_d1, 256, 512);
    k_transpose_bf16<<<dim3(1536), blk, 0, stream>>>(Wd2, wt_d2, 512, 768);
    k_f32_to_bf16<<<dim3(256), blk, 0, stream>>>(cb, cb_bf, NLEVELS * NCODES * EDIM);
    k_code_norms<<<dim3(4), blk, 0, stream>>>(cb, norms);
    // zero xq (8388608 f32) + adjacent lossAcc (4 f32)
    k_zero_f32<<<dim3(32769), blk, 0, stream>>>(xq, 8388612);

    // ---- encoder
    k_gemm<128, false, 1><<<dim3(4, 1024), blk, 0, stream>>>(
        x, wt_e0, be0, nullptr, act1_bf, NROWS, 512, 768, 1.f, 1);
    k_gemm<128, true, 1><<<dim3(2, 1024), blk, 0, stream>>>(
        act1_bf, wt_e1, be1, nullptr, act2_bf, NROWS, 256, 512, 1.f, 1);
    k_gemm<64, true, 2><<<dim3(1, 1024), blk, 0, stream>>>(
        act2_bf, wt_e2, be2, res, res_bf, NROWS, 64, 256, 1.f, 0);

    // ---- residual quantization: 4 levels
    for (int l = 0; l < NLEVELS; ++l) {
        k_gemm<128, true, 0><<<dim3(2, 1024), blk, 0, stream>>>(
            res_bf, cb_bf + (size_t)l * NCODES * EDIM, norms + l * NCODES,
            S, nullptr, NROWS, NCODES, EDIM, -2.f, 0);
        k_rq_update<<<dim3(NROWS / 8), blk, 0, stream>>>(
            S, cb + (size_t)l * NCODES * EDIM, res, res_bf, xq, lossAcc,
            out_idx, l);
    }
    k_f32_to_bf16<<<dim3(32768), blk, 0, stream>>>(xq, xq_bf, NROWS * EDIM);

    // ---- decoder (reuse activation buffers)
    k_gemm<128, true, 1><<<dim3(2, 1024), blk, 0, stream>>>(
        xq_bf, wt_d0, bd0, nullptr, act2_bf, NROWS, 256, 64, 1.f, 1);
    k_gemm<128, true, 1><<<dim3(4, 1024), blk, 0, stream>>>(
        act2_bf, wt_d1, bd1, nullptr, act1_bf, NROWS, 512, 256, 1.f, 1);
    k_gemm<128, true, 0><<<dim3(6, 1024), blk, 0, stream>>>(
        act1_bf, wt_d2, bd2, out, nullptr, NROWS, 768, 512, 1.f, 0);

    k_finalize_loss<<<dim3(1), dim3(1), 0, stream>>>(lossAcc, out_loss);
}
